// ModernBertAttentionWithoutRoPE_68075231641882
// MI455X (gfx1250) — matrix operand order repaired and checked by reading, not compiled, not run on hardware
//
#include <hip/hip_runtime.h>
#include <hip/hip_bf16.h>

// ---------------------------------------------------------------------------
// ModernBERT sliding-window attention (no RoPE) for gfx1250 / MI455X.
// BS=4, S=2048, D=768, H=12, HD=64, WINDOW=64.  All matmuls on
// v_wmma_f32_16x16x32_bf16 (fp32 accumulation), windowed attention never
// materializes the S x S score matrix.
// ---------------------------------------------------------------------------

#define S_LEN  2048
#define DMODEL 768
#define NH     12
#define HDIM   64
#define NBATCH 4
#define QTILES (S_LEN / 16)          // 128 query tiles per (b,h)

typedef __attribute__((ext_vector_type(16))) __bf16 v16bf;
typedef __attribute__((ext_vector_type(8)))  __bf16 v8bf;
typedef __attribute__((ext_vector_type(4)))  __bf16 v4bf;
typedef __attribute__((ext_vector_type(8)))  float  v8f;

union ABfrag { v16bf v; v8bf h[2]; };

__device__ __forceinline__ __bf16 f2bf(float f) {
  // round-to-nearest-even fp32 -> bf16
  unsigned u = __builtin_bit_cast(unsigned, f);
  unsigned r = (u + 0x7FFFu + ((u >> 16) & 1u)) >> 16;
  return __builtin_bit_cast(__bf16, (unsigned short)r);
}

// ---------------------------------------------------------------------------
// fp32 -> bf16 bulk convert (4 elems / thread, 16B load + 8B store)
// ---------------------------------------------------------------------------
__global__ void cvt_f32_bf16(const float* __restrict__ src,
                             __bf16* __restrict__ dst, int n) {
  int i = (blockIdx.x * blockDim.x + threadIdx.x) * 4;
  if (i + 3 < n) {
    float4 f = *(const float4*)(src + i);
    v4bf o;
    o[0] = f2bf(f.x); o[1] = f2bf(f.y); o[2] = f2bf(f.z); o[3] = f2bf(f.w);
    *(v4bf*)(dst + i) = o;
  }
}

// ---------------------------------------------------------------------------
// QKV projection: C[8192, 2304] = Hbf[8192,768] x Wqkv[2304,768]^T
// One wave -> 16(M) x 64(N) tile, K-loop 768 in steps of 32.
// Epilogue scatters into per-head Q[s][hd], K[s][hd], and V transposed [hd][s]
// (so attention B-operands are K-contiguous per column).
// ---------------------------------------------------------------------------
__global__ __launch_bounds__(128) void qkv_gemm_kernel(
    const __bf16* __restrict__ Hbf, const __bf16* __restrict__ Wbf,
    __bf16* __restrict__ Qw, __bf16* __restrict__ Kw, __bf16* __restrict__ Vt)
{
  const int wib  = threadIdx.x >> 5;
  const int wid  = blockIdx.x * 4 + wib;
  const int NB   = (3 * DMODEL) / 64;        // 36 column blocks
  const int mt   = wid / NB;
  const int nb   = wid % NB;
  const int lane = threadIdx.x & 31;
  const int half = lane >> 4;
  const int ln   = lane & 15;

  const __bf16* Arow = Hbf + (size_t)(mt * 16 + ln) * DMODEL;
  v8f acc[4] = {};
  for (int kc = 0; kc < DMODEL; kc += 32) {
    ABfrag a;                                     // A: row ln, K=16h..16h+15
    a.h[0] = *(const v8bf*)(Arow + kc + 16 * half);
    a.h[1] = *(const v8bf*)(Arow + kc + 16 * half + 8);
#pragma unroll
    for (int nt = 0; nt < 4; ++nt) {
      const __bf16* Brow = Wbf + (size_t)(nb * 64 + nt * 16 + ln) * DMODEL;
      v16bf bfr = *(const v16bf*)(Brow + kc + 16 * half);   // B: col ln, K=16h..16h+15
      acc[nt] = __builtin_amdgcn_wmma_f32_16x16x32_bf16(
          false, a.v, false, bfr, (short)0, acc[nt], false, false);
    }
  }

  const int c   = nb / NH;                   // 0=Q, 1=K, 2=V (64-wide head-aligned)
  const int h   = nb % NH;
  const int gm0 = mt * 16;
  const int b   = gm0 >> 11;                 // /2048
  const int s0  = (gm0 & (S_LEN - 1)) + 8 * half;
  if (c < 2) {
    __bf16* dst = (c == 0 ? Qw : Kw) + (size_t)(b * NH + h) * S_LEN * HDIM;
#pragma unroll
    for (int nt = 0; nt < 4; ++nt) {
      const int hd = nt * 16 + ln;
#pragma unroll
      for (int i = 0; i < 8; ++i)
        dst[(size_t)(s0 + i) * HDIM + hd] = f2bf(acc[nt][i]);
    }
  } else {
    // V transposed [hd][s]: per lane, 8 consecutive s -> one 16B store
#pragma unroll
    for (int nt = 0; nt < 4; ++nt) {
      const int hd = nt * 16 + ln;
      union { v8bf v; __bf16 e[8]; } o;
#pragma unroll
      for (int i = 0; i < 8; ++i) o.e[i] = f2bf(acc[nt][i]);
      *(v8bf*)(Vt + ((size_t)(b * NH + h) * HDIM + hd) * S_LEN + s0) = o.v;
    }
  }
}

// ---------------------------------------------------------------------------
// Windowed attention: one wave per (b, h, 16-query tile).
// Scores over 9 key tiles (144 keys covers q0-64 .. q0+79), analytic mask,
// softmax via 16-lane butterflies, P through LDS (C-layout -> A-layout),
// P x V with V^T giving contiguous B fragments.  Out-of-range key tiles are
// address-clamped; their P columns are exactly zero so they contribute 0.
// ---------------------------------------------------------------------------
__global__ __launch_bounds__(128) void attn_kernel(
    const __bf16* __restrict__ Qw, const __bf16* __restrict__ Kw,
    const __bf16* __restrict__ Vt, __bf16* __restrict__ AO)
{
  __shared__ __align__(32) __bf16 P[4][16][160];   // per-wave 16 x (144+16 zero pad)
  const int wib  = threadIdx.x >> 5;
  const int wid  = blockIdx.x * 4 + wib;
  const int qt   = wid & (QTILES - 1);
  const int h    = (wid >> 7) % NH;
  const int b    = wid / (QTILES * NH);
  const int lane = threadIdx.x & 31;
  const int half = lane >> 4;
  const int ln   = lane & 15;
  const size_t bh = (size_t)(b * NH + h);
  const __bf16* Qb = Qw + bh * S_LEN * HDIM;
  const __bf16* Kb = Kw + bh * S_LEN * HDIM;
  const __bf16* Vb = Vt + bh * HDIM * S_LEN;
  const int q0 = qt * 16;

  ABfrag qa[2];                                    // Q rows, d-chunks 0..31, 32..63
#pragma unroll
  for (int d2 = 0; d2 < 2; ++d2) {
    const __bf16* p = Qb + (size_t)(q0 + ln) * HDIM + d2 * 32;
    qa[d2].h[0] = *(const v8bf*)(p + 16 * half);
    qa[d2].h[1] = *(const v8bf*)(p + 16 * half + 8);
  }

  float sreg[9][8];
#pragma unroll
  for (int t = 0; t < 9; ++t) {
    const int kt  = qt - 4 + t;
    const int ktc = kt < 0 ? 0 : (kt > QTILES - 1 ? QTILES - 1 : kt);
    const __bf16* Krow = Kb + (size_t)(ktc * 16 + ln) * HDIM;  // B col ln = key row
    v16bf kb0 = *(const v16bf*)(Krow + 16 * half);
    v16bf kb1 = *(const v16bf*)(Krow + 32 + 16 * half);
    v8f c = {};
    c = __builtin_amdgcn_wmma_f32_16x16x32_bf16(false, qa[0].v, false, kb0, (short)0, c, false, false);
    c = __builtin_amdgcn_wmma_f32_16x16x32_bf16(false, qa[1].v, false, kb1, (short)0, c, false, false);
    const bool tv = (kt >= 0) && (kt < QTILES);
#pragma unroll
    for (int i = 0; i < 8; ++i) {
      const int q  = q0 + i + 8 * half;
      const int kk = kt * 16 + ln;
      const int d  = q - kk;
      sreg[t][i] = (tv && d <= 64 && d >= -64) ? c[i] * 0.125f : -1.0e30f;
    }
  }

  float mx[8], sm[8];
#pragma unroll
  for (int i = 0; i < 8; ++i) {
    float v = sreg[0][i];
#pragma unroll
    for (int t = 1; t < 9; ++t) v = fmaxf(v, sreg[t][i]);
#pragma unroll
    for (int o = 1; o < 16; o <<= 1) v = fmaxf(v, __shfl_xor(v, o, 32));
    mx[i] = v;
  }
#pragma unroll
  for (int i = 0; i < 8; ++i) {
    float s = 0.f;
#pragma unroll
    for (int t = 0; t < 9; ++t) { float e = __expf(sreg[t][i] - mx[i]); sreg[t][i] = e; s += e; }
#pragma unroll
    for (int o = 1; o < 16; o <<= 1) s += __shfl_xor(s, o, 32);
    sm[i] = s;
  }

  // P: C-layout -> LDS, zero pad cols 144..159
#pragma unroll
  for (int t = 0; t < 9; ++t)
#pragma unroll
    for (int i = 0; i < 8; ++i)
      P[wib][i + 8 * half][t * 16 + ln] = f2bf(sreg[t][i]);
#pragma unroll
  for (int i = 0; i < 8; ++i)
    P[wib][i + 8 * half][144 + ln] = f2bf(0.0f);

  const int win0 = q0 - 64;
  v8f acc[4] = {};
#pragma unroll
  for (int kc = 0; kc < 160; kc += 32) {
    ABfrag pa;                                     // A-layout reload of P from LDS
    pa.h[0] = *(const v8bf*)&P[wib][ln][kc + 16 * half];
    pa.h[1] = *(const v8bf*)&P[wib][ln][kc + 16 * half + 8];
    int off = win0 + kc + 16 * half;
    off = off < 0 ? 0 : (off > S_LEN - 16 ? S_LEN - 16 : off);  // clamped: P==0 there
#pragma unroll
    for (int dt = 0; dt < 4; ++dt) {
      v16bf vb = *(const v16bf*)(Vb + (size_t)(dt * 16 + ln) * S_LEN + off);
      acc[dt] = __builtin_amdgcn_wmma_f32_16x16x32_bf16(
          false, pa.v, false, vb, (short)0, acc[dt], false, false);
    }
  }

#pragma unroll
  for (int dt = 0; dt < 4; ++dt)
#pragma unroll
    for (int i = 0; i < 8; ++i) {
      const float o = acc[dt][i] / sm[i];
      AO[((size_t)b * S_LEN + q0 + i + 8 * half) * DMODEL + h * HDIM + dt * 16 + ln] = f2bf(o);
    }
}

// ---------------------------------------------------------------------------
// Output projection: Out[8192,768] (f32) = AO[8192,768] x Wo[768,768]^T
// ---------------------------------------------------------------------------
__global__ __launch_bounds__(128) void out_gemm_kernel(
    const __bf16* __restrict__ AO, const __bf16* __restrict__ Wobf,
    float* __restrict__ Out)
{
  const int wib  = threadIdx.x >> 5;
  const int wid  = blockIdx.x * 4 + wib;
  const int NB   = DMODEL / 64;              // 12
  const int mt   = wid / NB;
  const int nb   = wid % NB;
  const int lane = threadIdx.x & 31;
  const int half = lane >> 4;
  const int ln   = lane & 15;

  const __bf16* Arow = AO + (size_t)(mt * 16 + ln) * DMODEL;
  v8f acc[4] = {};
  for (int kc = 0; kc < DMODEL; kc += 32) {
    ABfrag a;
    a.h[0] = *(const v8bf*)(Arow + kc + 16 * half);
    a.h[1] = *(const v8bf*)(Arow + kc + 16 * half + 8);
#pragma unroll
    for (int nt = 0; nt < 4; ++nt) {
      const __bf16* Brow = Wobf + (size_t)(nb * 64 + nt * 16 + ln) * DMODEL;
      v16bf bfr = *(const v16bf*)(Brow + kc + 16 * half);
      acc[nt] = __builtin_amdgcn_wmma_f32_16x16x32_bf16(
          false, a.v, false, bfr, (short)0, acc[nt], false, false);
    }
  }
  const int gm0 = mt * 16 + 8 * half;
#pragma unroll
  for (int nt = 0; nt < 4; ++nt) {
    const int e = nb * 64 + nt * 16 + ln;
#pragma unroll
    for (int i = 0; i < 8; ++i)
      Out[(size_t)(gm0 + i) * DMODEL + e] = acc[nt][i];
  }
}

// ---------------------------------------------------------------------------
extern "C" void kernel_launch(void* const* d_in, const int* in_sizes, int n_in,
                              void* d_out, int out_size, void* d_ws, size_t ws_size,
                              hipStream_t stream) {
  const float* hidden = (const float*)d_in[0];   // [4,2048,768]
  const float* Wqkv   = (const float*)d_in[1];   // [2304,768]
  const float* Wo     = (const float*)d_in[2];   // [768,768]
  // d_in[3] = sliding_window_mask: recomputed analytically, not read.
  float* out = (float*)d_out;

  char* ws = (char*)d_ws;
  size_t off = 0;
  __bf16* Hbf    = (__bf16*)(ws + off); off += (size_t)NBATCH * S_LEN * DMODEL * 2;
  __bf16* Wqkvbf = (__bf16*)(ws + off); off += (size_t)3 * DMODEL * DMODEL * 2;
  __bf16* Wobf   = (__bf16*)(ws + off); off += (size_t)DMODEL * DMODEL * 2;
  __bf16* Qw     = (__bf16*)(ws + off); off += (size_t)NBATCH * NH * S_LEN * HDIM * 2;
  __bf16* Kw     = (__bf16*)(ws + off); off += (size_t)NBATCH * NH * S_LEN * HDIM * 2;
  __bf16* Vt     = (__bf16*)(ws + off); off += (size_t)NBATCH * NH * HDIM * S_LEN * 2;
  __bf16* AO     = (__bf16*)(ws + off); off += (size_t)NBATCH * S_LEN * DMODEL * 2;
  (void)ws_size; (void)in_sizes; (void)n_in; (void)out_size;

  const int nH  = NBATCH * S_LEN * DMODEL;   // 6291456
  const int nW1 = 3 * DMODEL * DMODEL;       // 1769472
  const int nW2 = DMODEL * DMODEL;           // 589824
  cvt_f32_bf16<<<nH  / 4 / 256, 256, 0, stream>>>(hidden, Hbf, nH);
  cvt_f32_bf16<<<nW1 / 4 / 256, 256, 0, stream>>>(Wqkv, Wqkvbf, nW1);
  cvt_f32_bf16<<<nW2 / 4 / 256, 256, 0, stream>>>(Wo, Wobf, nW2);

  // QKV GEMM: 512 M-tiles x 36 N-blocks = 18432 waves / 4 per block
  qkv_gemm_kernel<<<(512 * 36) / 4, 128, 0, stream>>>(Hbf, Wqkvbf, Qw, Kw, Vt);
  // Attention: 4*12*128 = 6144 waves
  attn_kernel<<<(NBATCH * NH * QTILES) / 4, 128, 0, stream>>>(Qw, Kw, Vt, AO);
  // Output projection: 512 x 12 = 6144 waves
  out_gemm_kernel<<<(512 * 12) / 4, 128, 0, stream>>>(AO, Wobf, out);
}
